// NetVLAD_24996709663159
// MI455X (gfx1250) — matrix operand order repaired and checked by reading, not compiled, not run on hardware
//
#include <hip/hip_runtime.h>

// ---------------------------------------------------------------------------
// NetVLAD forward, MI455X (gfx1250): bf16 WMMA 16x16x32 with hi/lo split-
// precision fp32 emulation + Tensor Data Mover (TDM) async x-tile staging.
// ---------------------------------------------------------------------------

typedef __attribute__((ext_vector_type(16))) __bf16 v16bf;
typedef __attribute__((ext_vector_type(8)))  float  v8f;
typedef __attribute__((ext_vector_type(4)))  unsigned int u32x4;
typedef __attribute__((ext_vector_type(8)))  int i32x8;
typedef __attribute__((ext_vector_type(4)))  int i32x4;

#define KC   64      // clusters
#define DD   512     // channels
#define NN   4096    // spatial positions (64*64)
#define BB   16      // batch
#define XSTR 520     // padded LDS row stride (bf16) for x^T  (n-major) tiles
#define NSTR 40      // padded LDS row stride (bf16) for x    (d-major) tiles
#define WSTR 40      // padded LDS row stride (bf16) for w tiles
#define EPSN 1e-12f

// LDS layout (bytes):
//   xf32  : [512][32] f32 TDM staging          65536
//   xThi  : [32][XSTR] bf16                    33280
//   xTlo  : [32][XSTR] bf16                    33280
//   xdnhi : [512][NSTR] bf16                   40960
//   xdnlo : [512][NSTR] bf16                   40960
//   lg    : [64][32] f32                        8192
//   whi   : [64][WSTR] bf16                     5120
//   wlo   : [64][WSTR] bf16                     5120
#define SMEM_BYTES 232448

union V16U { v16bf v; __bf16 e[16]; };

static __device__ __forceinline__ v16bf load16(const __bf16* p0, const __bf16* p1) {
  V16U u;
  __builtin_memcpy(&u.e[0], p0, 16);   // 8 bf16, 16B aligned -> b128
  __builtin_memcpy(&u.e[8], p1, 16);
  return u.v;
}

static __device__ __forceinline__ v8f wmma_bf16(v16bf a, v16bf b, v8f c) {
  return __builtin_amdgcn_wmma_f32_16x16x32_bf16(
      /*neg_a=*/false, a, /*neg_b=*/false, b,
      /*c_mod=*/(short)0, c, /*reuse_a=*/false, /*reuse_b=*/false);
}

// Issue a TDM load of x[b][0:512][n0:n0+32] (fp32, row-major, stride NN)
// into LDS staging buffer at wave-relative byte address lds_addr.
static __device__ __forceinline__ void tdm_load_tile(const float* x, int b, int n0,
                                                     unsigned lds_addr) {
  unsigned long long ga = (unsigned long long)(size_t)x
                        + ((unsigned long long)b * DD) * NN * 4ull
                        + (unsigned long long)n0 * 4ull;
  u32x4 g0;
  g0[0] = 1u;                                            // count=1, user mode
  g0[1] = lds_addr;                                      // LDS byte address
  g0[2] = (unsigned)(ga & 0xFFFFFFFFull);                // global addr [31:0]
  g0[3] = (unsigned)((ga >> 32) & 0x01FFFFFFull)         // global addr [56:32]
        | (2u << 30);                                    // type = 2 ("image")
  i32x8 g1;
  g1[0] = (int)(2u << 16);          // workgroup_mask=0, data_size=2 (4 bytes)
  g1[1] = (int)((unsigned)NN << 16);     // tensor_dim0 = 4096 (n)
  g1[2] = (int)((unsigned)DD << 16);     // tensor_dim1 = 512  (d)
  g1[3] = (int)(32u << 16);              // tile_dim0 = 32 (n, contiguous)
  g1[4] = 512;                           // tile_dim1 = 512 (d rows)
  g1[5] = NN;                            // tensor_dim0_stride = 4096
  g1[6] = 0;
  g1[7] = 0;
  i32x4 z4 = {0, 0, 0, 0};               // groups 2/3 unused (2-D tensor)
  i32x8 z8 = {0, 0, 0, 0, 0, 0, 0, 0};
  // 6-arg form (this toolchain): (g0, g1, g2, g3, extra, cpol)
  __builtin_amdgcn_tensor_load_to_lds(g0, g1, z4, z4, z8, 0);
}

// ---------------------------------------------------------------------------
// Kernel 0: zero accumulators, split conv_w into bf16 hi/lo.
// ---------------------------------------------------------------------------
__global__ void netvlad_init(const float* __restrict__ conv_w,
                             float* wx, float* wsum, float* gsum,
                             __bf16* cwhi, __bf16* cwlo) {
  int idx = blockIdx.x * 256 + threadIdx.x;
  wx[idx] = 0.0f;                                  // 16*64*512 = 524288
  if (idx < KC * DD) {                             // 32768
    float v = conv_w[idx];
    __bf16 h = (__bf16)v;
    cwhi[idx] = h;
    cwlo[idx] = (__bf16)(v - (float)h);
  }
  if (idx < BB * KC) wsum[idx] = 0.0f;             // 1024
  if (idx < BB)      gsum[idx] = 0.0f;             // 16
}

// ---------------------------------------------------------------------------
// Kernel 1: fused logits -> softmax -> (wx, wsum).  One HBM pass over x.
// grid = B*32 = 512 workgroups, 256 threads (8 wave32).  TDM pipelined:
// tile it+1 DMAs while tile it is computed.
// ---------------------------------------------------------------------------
__global__ __launch_bounds__(256, 1)
void netvlad_main(const float* __restrict__ x,
                  const __bf16* __restrict__ cwhi,
                  const __bf16* __restrict__ cwlo,
                  const float* __restrict__ conv_b,
                  float* __restrict__ wx,
                  float* __restrict__ wsum) {
  extern __shared__ char smem[];
  float*  xf32  = (float*)smem;                        // [512][32] staging
  __bf16* xThi  = (__bf16*)(smem + 65536);             // [32][XSTR]
  __bf16* xTlo  = xThi + 32 * XSTR;
  __bf16* xdnhi = xTlo + 32 * XSTR;                    // [512][NSTR]
  __bf16* xdnlo = xdnhi + DD * NSTR;
  float*  lg    = (float*)(xdnlo + DD * NSTR);         // [64][32]
  __bf16* whi   = (__bf16*)(lg + KC * 32);             // [64][WSTR]
  __bf16* wlo   = whi + KC * WSTR;

  const int b     = blockIdx.x >> 5;
  const int chunk = blockIdx.x & 31;
  const int nbase = chunk * 128;
  const int tid   = threadIdx.x;
  const int lane  = tid & 31;
  const int wid   = tid >> 5;
  const int half  = lane >> 4;
  const int l15   = lane & 15;

  // dynamic LDS begins right after static LDS (none here)
  const unsigned lds_stage = (unsigned)__builtin_amdgcn_groupstaticsize();

  v8f acc2[4][4];
#pragma unroll
  for (int mt = 0; mt < 4; ++mt)
#pragma unroll
    for (int dt = 0; dt < 4; ++dt)
      acc2[mt][dt] = (v8f){0.f, 0.f, 0.f, 0.f, 0.f, 0.f, 0.f, 0.f};

  if (wid == 0) tdm_load_tile(x, b, nbase, lds_stage);   // prologue DMA

  for (int it = 0; it < 4; ++it) {
    if (wid == 0) __builtin_amdgcn_s_wait_tensorcnt(0);
    __syncthreads();   // staging tile ready; previous iteration fully drained

    // ---- Stage 1: convert staged fp32 tile -> bf16 hi/lo in two layouts.
    {
#pragma unroll 4
      for (int s = 0; s < 32; ++s) {
        int p  = tid + s * 256;        // 8192 (d, n-pair) units
        int d  = p >> 4;
        int np = p & 15;
        float2 v = ((const float2*)(xf32 + d * 32))[np];
        __bf16 h0 = (__bf16)v.x, h1 = (__bf16)v.y;
        __bf16 l0 = (__bf16)(v.x - (float)h0), l1 = (__bf16)(v.y - (float)h1);
        int n = np * 2;
        xThi[(n    ) * XSTR + d] = h0;    // n-major (GEMM1 B operand)
        xThi[(n + 1) * XSTR + d] = h1;
        xTlo[(n    ) * XSTR + d] = l0;
        xTlo[(n + 1) * XSTR + d] = l1;
        union { __bf16 h[2]; unsigned u; } ph, pl;
        ph.h[0] = h0; ph.h[1] = h1;
        pl.h[0] = l0; pl.h[1] = l1;
        ((unsigned*)(xdnhi + d * NSTR))[np] = ph.u;   // d-major (GEMM2 B)
        ((unsigned*)(xdnlo + d * NSTR))[np] = pl.u;
      }
    }
    __syncthreads();   // conversion done: staging free, bf16 tiles ready

    // ---- Kick DMA for next tile; it overlaps GEMM1/softmax/GEMM2 below.
    if (wid == 0 && it < 3)
      tdm_load_tile(x, b, nbase + (it + 1) * 32, lds_stage);

    // ---- Stage 2: GEMM1  logits(64x32) = conv_w(64x512) * x(512x32)
    {
      const int mt = wid >> 1, nt = wid & 1;
      const int m  = mt * 16 + l15;
      v8f acc = (v8f){0.f, 0.f, 0.f, 0.f, 0.f, 0.f, 0.f, 0.f};
#pragma unroll 4
      for (int d0 = 0; d0 < DD; d0 += 32) {
        int ab = m * DD + d0 + half * 8;                  // A: k=(j&8)*2+(j&7)+half*8
        v16bf ahi = load16(cwhi + ab, cwhi + ab + 16);
        v16bf alo = load16(cwlo + ab, cwlo + ab + 16);
        int bb = (nt * 16 + l15) * XSTR + d0 + half * 8;  // B: k=(j&8)*2+(j&7)+half*8
        v16bf bhi = load16(xThi + bb, xThi + bb + 16);
        v16bf blo = load16(xTlo + bb, xTlo + bb + 16);
        acc = wmma_bf16(ahi, bhi, acc);
        acc = wmma_bf16(ahi, blo, acc);
        acc = wmma_bf16(alo, bhi, acc);
      }
#pragma unroll
      for (int r = 0; r < 8; ++r)       // C: m=r+half*8, n=lane&15
        lg[(mt * 16 + half * 8 + r) * 32 + nt * 16 + l15] = acc[r];
    }
    __syncthreads();

    // ---- Stage 3: softmax over 64 clusters per column (threads 0..31).
    if (tid < 32) {
      const int n = tid;
      float mx = -3.402823466e+38f;
      for (int k = 0; k < KC; ++k) {
        float l = lg[k * 32 + n] + conv_b[k];
        lg[k * 32 + n] = l;
        mx = fmaxf(mx, l);
      }
      float s = 0.f;
      for (int k = 0; k < KC; ++k) {
        float e = __expf(lg[k * 32 + n] - mx);
        lg[k * 32 + n] = e;
        s += e;
      }
      float inv = 1.f / s;
      for (int k = 0; k < KC; ++k) {
        float w = lg[k * 32 + n] * inv;
        lg[k * 32 + n] = w;
        __bf16 h = (__bf16)w;
        whi[k * WSTR + n] = h;
        wlo[k * WSTR + n] = (__bf16)(w - (float)h);
      }
    }
    __syncthreads();

    // ---- Stage 4a: wsum[k] += sum_n w[k][n]  (threads 0..63)
    if (tid < KC) {
      float s = 0.f;
#pragma unroll 8
      for (int n = 0; n < 32; ++n) s += lg[tid * 32 + n];
      unsafeAtomicAdd(&wsum[b * KC + tid], s);
    }

    // ---- Stage 4b: GEMM2  wx(64x512) += w(64x32) * x^T(32x512)
#pragma unroll
    for (int dt = 0; dt < 4; ++dt) {
      const int dcol = (wid * 4 + dt) * 16 + l15;
      const int bb2  = dcol * NSTR + half * 8;            // B: row n=(j&8)*2+(j&7)+half*8
      v16bf bh = load16(xdnhi + bb2, xdnhi + bb2 + 16);
      v16bf bl = load16(xdnlo + bb2, xdnlo + bb2 + 16);
#pragma unroll
      for (int mt = 0; mt < 4; ++mt) {
        int ab = (mt * 16 + l15) * WSTR + half * 8;       // A layout
        v16bf awhi = load16(whi + ab, whi + ab + 16);
        v16bf awlo = load16(wlo + ab, wlo + ab + 16);
        v8f c = acc2[mt][dt];
        c = wmma_bf16(awhi, bh, c);
        c = wmma_bf16(awhi, bl, c);
        c = wmma_bf16(awlo, bh, c);
        acc2[mt][dt] = c;
      }
    }
  }

  // ---- Flush persistent wx accumulators with f32 atomics (L2-resident).
#pragma unroll
  for (int mt = 0; mt < 4; ++mt)
#pragma unroll
    for (int dt = 0; dt < 4; ++dt)
#pragma unroll
      for (int r = 0; r < 8; ++r) {
        int kc = mt * 16 + half * 8 + r;
        int d  = (wid * 4 + dt) * 16 + l15;
        unsafeAtomicAdd(&wx[((size_t)b * KC + kc) * DD + d], acc2[mt][dt][r]);
      }
}

// ---------------------------------------------------------------------------
// Kernel 2: vlad = wx - wsum*centers; intra-normalize each (b,k) row of 512.
// ---------------------------------------------------------------------------
__global__ void netvlad_finalize1(const float* __restrict__ wx,
                                  const float* __restrict__ wsum,
                                  const float* __restrict__ centers,
                                  float* __restrict__ out,
                                  float* __restrict__ gsum) {
  const int bk = blockIdx.x;
  const int b = bk >> 6, k = bk & 63;
  const int tid = threadIdx.x;
  const float ws = wsum[b * KC + k];

  float v0 = wx[(size_t)bk * DD + tid]       - ws * centers[k * DD + tid];
  float v1 = wx[(size_t)bk * DD + 256 + tid] - ws * centers[k * DD + 256 + tid];

  __shared__ float red[256];
  red[tid] = v0 * v0 + v1 * v1;
  __syncthreads();
  for (int s = 128; s > 0; s >>= 1) {
    if (tid < s) red[tid] += red[tid + s];
    __syncthreads();
  }
  const float tot   = red[0];
  const float norm  = sqrtf(tot);
  const float scale = 1.0f / fmaxf(norm, EPSN);

  out[(size_t)b * (KC * DD) + k * DD + tid]       = v0 * scale;
  out[(size_t)b * (KC * DD) + k * DD + 256 + tid] = v1 * scale;
  if (tid == 0) unsafeAtomicAdd(&gsum[b], tot * scale * scale);
}

// ---------------------------------------------------------------------------
// Kernel 3: global L2 normalize per batch image.
// ---------------------------------------------------------------------------
__global__ void netvlad_finalize2(float* __restrict__ out,
                                  const float* __restrict__ gsum) {
  int idx = blockIdx.x * 256 + threadIdx.x;
  int b = idx >> 15;
  float g = gsum[b];
  out[idx] *= 1.0f / fmaxf(sqrtf(g), EPSN);
}

// ---------------------------------------------------------------------------
extern "C" void kernel_launch(void* const* d_in, const int* in_sizes, int n_in,
                              void* d_out, int out_size, void* d_ws, size_t ws_size,
                              hipStream_t stream) {
  (void)in_sizes; (void)n_in; (void)out_size; (void)ws_size;

  const float* x       = (const float*)d_in[0];  // (16, 512, 64, 64)
  const float* conv_w  = (const float*)d_in[1];  // (64, 512)
  const float* conv_b  = (const float*)d_in[2];  // (64,)
  const float* centers = (const float*)d_in[3];  // (64, 512)
  float* out = (float*)d_out;                    // (16, 32768)

  float*  wx   = (float*)d_ws;                   // 524288 f32 (2 MB, L2)
  float*  wsum = wx + (size_t)BB * KC * DD;      // 1024 f32
  float*  gsum = wsum + BB * KC;                 // 16 f32
  __bf16* cwhi = (__bf16*)(gsum + BB);           // 32768 bf16
  __bf16* cwlo = cwhi + KC * DD;                 // 32768 bf16

  netvlad_init<<<2048, 256, 0, stream>>>(conv_w, wx, wsum, gsum, cwhi, cwlo);
  netvlad_main<<<BB * 32, 256, SMEM_BYTES, stream>>>(x, cwhi, cwlo, conv_b, wx, wsum);
  netvlad_finalize1<<<BB * KC, 256, 0, stream>>>(wx, wsum, centers, out, gsum);
  netvlad_finalize2<<<2048, 256, 0, stream>>>(out, gsum);
}
